// AttentionEncoder_37830071943234
// MI455X (gfx1250) — matrix operand order repaired and checked by reading, not compiled, not run on hardware
//
#include <hip/hip_runtime.h>
#include <hip/hip_bf16.h>
#include <stddef.h>

// ---------------------------------------------------------------------------
// CDNA5 (gfx1250) GRU encoder.
//   B=64, T=512, V=32000, E=512, H=1024, 3H=3072.
//   d_out = [enc_out (B*T*H) | enc_states (4*B*H) | pad_mask (B*T)]  (float)
// Strategy:
//   1) embedding gather -> bf16 activations, pad mask.
//   2) one-time f32->bf16 transposed weight copies (WT is [N,K] so WMMA
//      B-fragments are contiguous per lane).
//   3) per layer: big WMMA GEMM  XG = X @ Wx + bx   (bulk parallel)
//   4) per timestep: fused WMMA step kernel: gh = h @ Wh, gates, h update.
//      Wh^T bf16 (6.3MB) is L2-resident across all 512 steps.
// Workspace: ~585 MB (WxT/WhT bf16 ~47MB, two activation buffers 2*67MB,
// XG f32 403MB, h state buffers <1MB).
// ---------------------------------------------------------------------------

typedef __attribute__((ext_vector_type(16))) __bf16 v16bf;
typedef __attribute__((ext_vector_type(8)))  __bf16 v8bf;
typedef __attribute__((ext_vector_type(8)))  float  v8f;

// ---- WMMA helpers (layouts per CDNA5 ISA 7.12.2) ----------------

static __device__ __forceinline__ v8f wmma_bf16(v16bf a, v16bf b, v8f c) {
    // D = A(16x32 bf16) * B(32x16 bf16) + C(16x16 f32)
    return __builtin_amdgcn_wmma_f32_16x16x32_bf16(
        /*neg_a=*/false, a, /*neg_b=*/false, b,
        /*c_mod=*/(short)0, c, /*reuse_a=*/false, /*reuse_b=*/false);
}

// A fragment: row-major A [M x K], tile at (m0, k0).
// lane<16: row m0+lane,    elems i -> K=k0+i
// lane>=16: row m0+lane-16, elems i -> K=k0+16+i
static __device__ __forceinline__ v16bf load_a_frag(const __bf16* __restrict__ A,
                                                    int lda, int m0, int k0,
                                                    int l16, int hi) {
    const __bf16* p = A + (size_t)(m0 + l16) * lda + k0 + (hi ? 16 : 0);
    union { v16bf v; v8bf h[2]; } u;
    u.h[0] = *(const v8bf*)(p);        // 16B contiguous
    u.h[1] = *(const v8bf*)(p + 8);    // 16B contiguous
    return u.v;
}

// B fragment from W^T stored [N x K] row-major: tile cols n0..n0+15, k0..k0+31.
// lane<16: col n0+lane,     elems i -> K=k0+i      (32B contiguous)
// lane>=16: col n0+lane-16, elems i -> K=k0+16+i   (32B contiguous)
static __device__ __forceinline__ v16bf load_b_frag(const __bf16* __restrict__ WT,
                                                    int ldk, int n0, int k0,
                                                    int l16, int hi) {
    const __bf16* p = WT + (size_t)(n0 + l16) * ldk + k0 + (hi ? 16 : 0);
    return *(const v16bf*)(p);
}

static __device__ __forceinline__ float sigmoidf_(float x) {
    return 1.0f / (1.0f + __expf(-x));
}

// ---- 1) embedding gather + pad mask ---------------------------------------
__global__ void __launch_bounds__(256)
embed_gather(const int* __restrict__ x, const float* __restrict__ emb,
             __bf16* __restrict__ X, float* __restrict__ mask, int E) {
    int row = blockIdx.x;                 // row = b*T + t
    int tok = x[row];
    if (threadIdx.x == 0) mask[row] = (tok != 0) ? 1.0f : 0.0f;
    const float* src = emb + (size_t)tok * E;
    __bf16* dst = X + (size_t)row * E;
    for (int e = threadIdx.x; e < E; e += blockDim.x)
        dst[e] = (__bf16)src[e];
}

// ---- 2) f32 [K,N] -> bf16 [N,K] transpose-convert (one-time) --------------
__global__ void __launch_bounds__(256)
transpose_cvt(const float* __restrict__ in, __bf16* __restrict__ out,
              int K, int N) {
    int i = blockIdx.x * blockDim.x + threadIdx.x;
    if (i >= K * N) return;
    int n = i / K, k = i - n * K;
    out[(size_t)n * K + k] = (__bf16)in[(size_t)k * N + n];
}

__global__ void __launch_bounds__(256)
zero_state(float* __restrict__ hf, __bf16* __restrict__ hb, int n) {
    int i = blockIdx.x * blockDim.x + threadIdx.x;
    if (i < n) { hf[i] = 0.0f; hb[i] = (__bf16)0.0f; }
}

// ---- 3) XG = X @ Wx + bx  (M x N, K-reduction via WMMA) --------------------
// Block = 256 thr = 8 waves; block tile 64 rows x 256 cols; wave tile 32x64.
__global__ void __launch_bounds__(256)
gemm_xg(const __bf16* __restrict__ X, const __bf16* __restrict__ WT,
        const float* __restrict__ bx, float* __restrict__ XG,
        int K, int N) {
    int tid  = threadIdx.x;
    int lane = tid & 31, w = tid >> 5;
    int l16  = lane & 15, hi = (lane >> 4) & 1;
    int m0 = blockIdx.y * 64 + (w & 1) * 32;
    int n0 = blockIdx.x * 256 + (w >> 1) * 64;

    v8f c[2][4] = {};
    for (int k0 = 0; k0 < K; k0 += 32) {
        v16bf a0 = load_a_frag(X, K, m0,      k0, l16, hi);
        v16bf a1 = load_a_frag(X, K, m0 + 16, k0, l16, hi);
#pragma unroll
        for (int ct = 0; ct < 4; ++ct) {
            v16bf b = load_b_frag(WT, K, n0 + ct * 16, k0, l16, hi);
            c[0][ct] = wmma_bf16(a0, b, c[0][ct]);
            c[1][ct] = wmma_bf16(a1, b, c[1][ct]);
        }
    }
#pragma unroll
    for (int ct = 0; ct < 4; ++ct) {
        int j = n0 + ct * 16 + l16;
        float bias = bx[j];
#pragma unroll
        for (int rt = 0; rt < 2; ++rt) {
#pragma unroll
            for (int v = 0; v < 8; ++v) {
                int row = m0 + rt * 16 + v + hi * 8;
                XG[(size_t)row * N + j] = c[rt][ct][v] + bias;
            }
        }
    }
}

// ---- 4) fused GRU timestep -------------------------------------------------
// gh = h_prev @ Wh  (via WhT bf16), then gate math + state update, fused.
// 256 waves total: 4 row-tiles (B=64) x 64 col-tiles (H=1024). 32 blocks.
// Each wave accumulates the SAME 16-col group in the r, z and n regions so it
// can finish the gates locally.
__global__ void __launch_bounds__(256)
gru_step(const __bf16* __restrict__ WhT,   // [3H, H] bf16
         const float* __restrict__ bh,     // [3H]
         const float* __restrict__ XG,     // [B*T, 3H] (includes bx)
         const float* __restrict__ Hf_in, const __bf16* __restrict__ Hb_in,
         float* __restrict__ Hf_out, __bf16* __restrict__ Hb_out,
         __bf16* __restrict__ Ybf,         // next-layer input (layers 0..2)
         float*  __restrict__ Yf,          // enc_out (layer 3)
         float*  __restrict__ hT_out,      // enc_states slot for this layer
         int t, int T, int H) {
    const int H3 = 3 * H;
    int tid  = threadIdx.x;
    int lane = tid & 31, l16 = lane & 15, hi = (lane >> 4) & 1;
    int w  = blockIdx.x * (blockDim.x >> 5) + (tid >> 5);
    int m0 = (w >> 6) * 16;        // batch tile
    int g0 = (w & 63) * 16;        // hidden-column tile

    v8f cr = {}, cz = {}, cn = {};
    for (int k0 = 0; k0 < H; k0 += 32) {
        v16bf a  = load_a_frag(Hb_in, H, m0, k0, l16, hi);
        v16bf br = load_b_frag(WhT, H, g0,         k0, l16, hi);
        v16bf bz = load_b_frag(WhT, H, H + g0,     k0, l16, hi);
        v16bf bn = load_b_frag(WhT, H, 2 * H + g0, k0, l16, hi);
        cr = wmma_bf16(a, br, cr);
        cz = wmma_bf16(a, bz, cz);
        cn = wmma_bf16(a, bn, cn);
    }

    int j = g0 + l16;
    float bhr = bh[j], bhz = bh[H + j], bhn = bh[2 * H + j];
#pragma unroll
    for (int v = 0; v < 8; ++v) {
        int b = m0 + v + hi * 8;                  // batch index (row)
        size_t xrow = (size_t)b * T + t;
        float xr = XG[xrow * H3 + j];
        float xz = XG[xrow * H3 + H + j];
        float xn = XG[xrow * H3 + 2 * H + j];
        float r  = sigmoidf_(xr + cr[v] + bhr);
        float z  = sigmoidf_(xz + cz[v] + bhz);
        float nn = tanhf(xn + r * (cn[v] + bhn));
        float hp = Hf_in[(size_t)b * H + j];
        float hn = (1.0f - z) * nn + z * hp;
        Hf_out[(size_t)b * H + j] = hn;
        Hb_out[(size_t)b * H + j] = (__bf16)hn;
        if (Ybf) Ybf[xrow * H + j] = (__bf16)hn;
        if (Yf)  Yf[xrow * H + j]  = hn;
        if (t == T - 1) hT_out[(size_t)b * H + j] = hn;
    }
}

// ---------------------------------------------------------------------------
extern "C" void kernel_launch(void* const* d_in, const int* in_sizes, int n_in,
                              void* d_out, int out_size, void* d_ws, size_t ws_size,
                              hipStream_t stream) {
    (void)in_sizes; (void)n_in; (void)out_size; (void)ws_size;
    const int B = 64, T = 512, E = 512, H = 1024, H3 = 3 * H;
    const int M = B * T;

    const int*   x     = (const int*)  d_in[0];
    const float* emb   = (const float*)d_in[1];
    const float* Wx0   = (const float*)d_in[2];   // [E, 3H]
    const float* Wh0   = (const float*)d_in[3];   // [H, 3H]
    const float* bx0   = (const float*)d_in[4];
    const float* bh0   = (const float*)d_in[5];
    const float* WxL   = (const float*)d_in[6];   // [3, H, 3H]
    const float* WhL   = (const float*)d_in[7];   // [3, H, 3H]
    const float* bxL   = (const float*)d_in[8];   // [3, 3H]
    const float* bhL   = (const float*)d_in[9];   // [3, 3H]

    float* out        = (float*)d_out;
    float* enc_out    = out;                               // [B*T*H]
    float* enc_states = out + (size_t)M * H;               // [4*B*H]
    float* mask       = enc_states + (size_t)4 * B * H;    // [B*T]

    // workspace carve-out
    char* ws = (char*)d_ws;
    size_t off = 0;
    auto take = [&](size_t bytes) -> void* {
        void* p = ws + off;
        off = (off + bytes + 255) & ~(size_t)255;
        return p;
    };
    __bf16* WxT[4];
    WxT[0] = (__bf16*)take((size_t)H3 * E * sizeof(__bf16));
    for (int l = 1; l < 4; ++l) WxT[l] = (__bf16*)take((size_t)H3 * H * sizeof(__bf16));
    __bf16* WhT[4];
    for (int l = 0; l < 4; ++l) WhT[l] = (__bf16*)take((size_t)H3 * H * sizeof(__bf16));
    __bf16* Xa  = (__bf16*)take((size_t)M * H * sizeof(__bf16));
    __bf16* Xb  = (__bf16*)take((size_t)M * H * sizeof(__bf16));
    float*  XG  = (float*) take((size_t)M * H3 * sizeof(float));
    float*  Hf0 = (float*) take((size_t)B * H * sizeof(float));
    float*  Hf1 = (float*) take((size_t)B * H * sizeof(float));
    __bf16* Hb0 = (__bf16*)take((size_t)B * H * sizeof(__bf16));
    __bf16* Hb1 = (__bf16*)take((size_t)B * H * sizeof(__bf16));

    // 1) embedding + pad mask
    embed_gather<<<M, 256, 0, stream>>>(x, emb, Xa, mask, E);

    // 2) weight transpose-convert (f32 [K,3H] -> bf16 [3H,K])
    {
        int n0 = H3 * E, n1 = H3 * H;
        transpose_cvt<<<(n0 + 255) / 256, 256, 0, stream>>>(Wx0, WxT[0], E, H3);
        for (int l = 0; l < 3; ++l)
            transpose_cvt<<<(n1 + 255) / 256, 256, 0, stream>>>(
                WxL + (size_t)l * H * H3, WxT[l + 1], H, H3);
        transpose_cvt<<<(n1 + 255) / 256, 256, 0, stream>>>(Wh0, WhT[0], H, H3);
        for (int l = 0; l < 3; ++l)
            transpose_cvt<<<(n1 + 255) / 256, 256, 0, stream>>>(
                WhL + (size_t)l * H * H3, WhT[l + 1], H, H3);
    }

    // 3) layers
    __bf16* Xcur = Xa;
    __bf16* Xnext = Xb;
    for (int l = 0; l < 4; ++l) {
        const int K = (l == 0) ? E : H;
        const float* bxl = (l == 0) ? bx0 : bxL + (size_t)(l - 1) * H3;
        const float* bhl = (l == 0) ? bh0 : bhL + (size_t)(l - 1) * H3;

        dim3 g(H3 / 256, M / 64);                       // (12, 512)
        gemm_xg<<<g, 256, 0, stream>>>(Xcur, WxT[l], bxl, XG, K, H3);

        zero_state<<<(B * H + 255) / 256, 256, 0, stream>>>(Hf0, Hb0, B * H);

        float* hTdst = enc_states + (size_t)l * B * H;
        __bf16* ybf = (l < 3) ? Xnext : nullptr;
        float*  yf  = (l == 3) ? enc_out : nullptr;
        for (int t = 0; t < T; ++t) {
            float*  fin  = (t & 1) ? Hf1 : Hf0;
            float*  fout = (t & 1) ? Hf0 : Hf1;
            __bf16* bin  = (t & 1) ? Hb1 : Hb0;
            __bf16* bout = (t & 1) ? Hb0 : Hb1;
            gru_step<<<32, 256, 0, stream>>>(WhT[l], bhl, XG, fin, bin,
                                             fout, bout, ybf, yf, hTdst,
                                             t, T, H);
        }
        __bf16* tmp = Xcur; Xcur = Xnext; Xnext = tmp;
    }
}
